// SplitCNN_1468878815381
// MI455X (gfx1250) — compile-verified, matrix operand order checked by reading, not run on hardware
//
#include <hip/hip_runtime.h>
#include <hip/hip_bf16.h>

typedef __attribute__((ext_vector_type(8)))  _Float16 v8h;
typedef __attribute__((ext_vector_type(16))) _Float16 v16h;
typedef __attribute__((ext_vector_type(8)))  float    v8f;

#define BATCH 1800
#define MPAD  1808   // 113 * 16
#define KTOT  4096   // 2*1024 + 8*256
#define NOUT  128

// ---------------------------------------------------------------- init ------
__global__ void k_init(float* chanstats, float* bstats, _Float16* A) {
  int t = blockIdx.x * 256 + threadIdx.x;
  if (t < 80) chanstats[t] = 0.f;
  if (t < 20) bstats[t] = 0.f;
  const int padElems = (MPAD - BATCH) * KTOT;
  for (int i = t; i < padElems; i += gridDim.x * 256)
    A[(size_t)BATCH * KTOT + i] = (_Float16)0.f;
}

// ------------------------------------------- conv (+stats / +BN,relu,pool) --
// pass 0: conv only, accumulate per-channel sum/sumsq (for BN over batch)
// pass 1: conv again, apply BN scale/shift, relu, 2x2 maxpool, store f16 to A,
//         accumulate per-branch sum/sumsq of pooled outputs (for cross-norm)
template <int TS>
__global__ void k_conv(const float* __restrict__ x,
                       const float* __restrict__ conv_w,   // [10][4][3][3][3]
                       const float* __restrict__ conv_b,   // [10][4]
                       const float* __restrict__ scaleshift,// [10][4][2]
                       float* __restrict__ chanstats,      // [10][4][2]
                       float* __restrict__ bstats,         // [10][2]
                       _Float16* __restrict__ A,           // [MPAD][KTOT]
                       int pass)
{
  constexpr int P = TS / 2;
  constexpr int CELLS = 4 * P * P;
  const int b  = blockIdx.x;
  const int br = blockIdx.y;
  int n, r0, c0, kbase;
  if (TS == 32) { n = br;     r0 = br * 32;        c0 = 0;                  kbase = br * 1024; }
  else          { n = 2 + br; r0 = (br & 3) * 16;  c0 = 32 + (br >> 2) * 16; kbase = 2048 + br * 256; }

  __shared__ float tile[3][TS + 2][TS + 2];   // zero-padded (SAME on the slice)
  __shared__ float wsm[4][27];
  __shared__ float bias_s[4], scl_s[4], sft_s[4];
  __shared__ float redS[4], redQ[4];

  const int t = threadIdx.x; // 256 threads
  for (int i = t; i < 3 * (TS + 2) * (TS + 2); i += 256) ((float*)tile)[i] = 0.f;
  if (t < 108) ((float*)wsm)[t] = conv_w[n * 108 + t];
  if (t < 4) {
    bias_s[t] = conv_b[n * 4 + t];
    scl_s[t]  = scaleshift[(n * 4 + t) * 2 + 0];  // unused (stale) in pass 0
    sft_s[t]  = scaleshift[(n * 4 + t) * 2 + 1];
    redS[t] = 0.f; redQ[t] = 0.f;
  }
  __syncthreads();
  for (int i = t; i < 3 * TS * TS; i += 256) {
    int ci = i / (TS * TS), rem = i % (TS * TS), ii = rem / TS, jj = rem % TS;
    tile[ci][ii + 1][jj + 1] =
        x[((size_t)(b * 3 + ci) * 64 + (r0 + ii)) * 64 + (c0 + jj)];
  }
  __syncthreads();

  for (int cell = t; cell < CELLS; cell += 256) {
    const int c  = cell / (P * P);
    const int rm = cell % (P * P);
    const int i0 = (rm / P) * 2, j0 = (rm % P) * 2;
    float v[4];
#pragma unroll
    for (int q = 0; q < 4; ++q) {
      const int oi = i0 + (q >> 1), oj = j0 + (q & 1);
      float s = bias_s[c];
#pragma unroll
      for (int ci = 0; ci < 3; ++ci)
#pragma unroll
        for (int di = 0; di < 3; ++di)
#pragma unroll
          for (int dj = 0; dj < 3; ++dj)
            s += tile[ci][oi + di][oj + dj] * wsm[c][ci * 9 + di * 3 + dj];
      v[q] = s;
    }
    if (pass == 0) {
      atomicAdd(&redS[c], v[0] + v[1] + v[2] + v[3]);
      atomicAdd(&redQ[c], v[0]*v[0] + v[1]*v[1] + v[2]*v[2] + v[3]*v[3]);
    } else {
      const float scl = scl_s[c], sft = sft_s[c];
      float m = scl * v[0] + sft;
      m = fmaxf(m, scl * v[1] + sft);
      m = fmaxf(m, scl * v[2] + sft);
      m = fmaxf(m, scl * v[3] + sft);
      m = fmaxf(m, 0.f);                       // relu(max(...)) == max(relu(...))
      A[(size_t)b * KTOT + kbase + cell] = (_Float16)m;
      atomicAdd(&redS[0], m);
      atomicAdd(&redQ[0], m * m);
    }
  }
  __syncthreads();
  if (pass == 0) {
    if (t < 4) {
      atomicAdd(&chanstats[(n * 4 + t) * 2 + 0], redS[t]);
      atomicAdd(&chanstats[(n * 4 + t) * 2 + 1], redQ[t]);
    }
  } else if (t == 0) {
    atomicAdd(&bstats[n * 2 + 0], redS[0]);
    atomicAdd(&bstats[n * 2 + 1], redQ[0]);
  }
}

// ------------------------------------------------- BN stat finalization -----
__global__ void k_chanfin(const float* chanstats, const float* bn_g,
                          const float* bn_b, float* scaleshift) {
  int idx = threadIdx.x;
  if (idx >= 40) return;
  int n = idx / 4;
  float cnt = (n < 2) ? (float)(BATCH * 32 * 32) : (float)(BATCH * 16 * 16);
  float mu  = chanstats[idx * 2 + 0] / cnt;
  float var = chanstats[idx * 2 + 1] / cnt - mu * mu;
  float scl = bn_g[idx] * rsqrtf(var + 1e-5f);
  scaleshift[idx * 2 + 0] = scl;
  scaleshift[idx * 2 + 1] = bn_b[idx] - mu * scl;
}

// ------------------------- fused weight build: Wall[j,k] = w2[n,j]*cutW -----
// also d[j] = noise_j - inv*mean*rowsum(Wall_j), misc[0] = inv
__global__ void k_weights(const int* __restrict__ assign_idx,
                          const float* __restrict__ h_mat,     // [10][128][5]
                          const float* __restrict__ b_mat,     // [10][128]
                          const float* __restrict__ a_arr,     // [128][5]
                          const float* __restrict__ noise,     // [128][5]
                          const float* __restrict__ cutW_big,  // [2][128][1024]
                          const float* __restrict__ cutW_small,// [8][128][256]
                          const float* __restrict__ bstats,    // [10][2]
                          _Float16* __restrict__ Wall,         // [128][KTOT]
                          float* __restrict__ dvec,
                          float* __restrict__ misc)
{
  const int j = blockIdx.x;       // output column 0..127
  const int t = threadIdx.x;      // 256
  __shared__ float w2s[10];
  __shared__ float sred;
  if (t == 0) sred = 0.f;
  if (t < 10) {
    const int ai = assign_idx[j];
    float hv = 0.f;
#pragma unroll
    for (int d = 0; d < 5; ++d)
      hv += a_arr[j * 5 + d] * h_mat[((size_t)t * 128 + ai) * 5 + d];
    w2s[t] = b_mat[t * 128 + j] * hv;
  }
  __syncthreads();
  float ls = 0.f;
  for (int k = t; k < KTOT; k += 256) {
    float wv; int n;
    if (k < 2048) {
      n = k >> 10;
      wv = cutW_big[(size_t)n * 128 * 1024 + (size_t)j * 1024 + (k & 1023)];
    } else {
      int i = (k - 2048) >> 8; n = 2 + i;
      wv = cutW_small[(size_t)i * 128 * 256 + (size_t)j * 256 + ((k - 2048) & 255)];
    }
    float val = w2s[n] * wv;
    Wall[(size_t)j * KTOT + k] = (_Float16)val;
    ls += val;
  }
  atomicAdd(&sred, ls);
  __syncthreads();
  if (t == 0) {
    float msum = 0.f, vsum = 0.f;
#pragma unroll
    for (int n = 0; n < 10; ++n) {
      float cnt = (n < 2) ? (float)(BATCH * 1024) : (float)(BATCH * 256);
      float bm = bstats[n * 2 + 0] / cnt;
      float bv = bstats[n * 2 + 1] / cnt - bm * bm;
      msum += bm; vsum += bv;
    }
    float mean = msum * 0.1f;
    float inv  = rsqrtf(vsum * 0.1f + 1e-6f);
    float cn = 0.f;
#pragma unroll
    for (int d = 0; d < 5; ++d) cn += a_arr[j * 5 + d] * noise[j * 5 + d];
    dvec[j] = cn - inv * mean * sred;
    misc[0] = inv;                        // same value from every block: benign
  }
}

// ------------------------------- WMMA GEMM: R = relu(inv*(A@Wall^T) + d) ----
// A [MPAD x KTOT] f16 row-major, Wall [NOUT x KTOT] f16 row-major (B^T).
// Fragment layout per ISA 7.12.2 (16-bit A 16x32): lanes 0-15 hold K 0..7 and
// 16..23; lanes 16-31 hold K 8..15 and 24..31. B assumed symmetric.
__global__ void k_gemm(const _Float16* __restrict__ A,
                       const _Float16* __restrict__ Wall,
                       const float* __restrict__ dvec,
                       const float* __restrict__ misc,
                       float* __restrict__ R)
{
  const int lane = threadIdx.x & 31;
  const int wave = threadIdx.x >> 5;         // 0..7 -> N tile
  const int m0 = blockIdx.x * 16;
  const int n0 = wave * 16;
  const int base = (lane < 16) ? 0 : 8;
  const _Float16* pA = A    + (size_t)(m0 + (lane & 15)) * KTOT + base;
  const _Float16* pB = Wall + (size_t)(n0 + (lane & 15)) * KTOT + base;
  v8f acc = {};
#pragma unroll 4
  for (int k0 = 0; k0 < KTOT; k0 += 32) {
    v8h a0 = *(const v8h*)(pA + k0);
    v8h a1 = *(const v8h*)(pA + k0 + 16);
    v8h b0 = *(const v8h*)(pB + k0);
    v8h b1 = *(const v8h*)(pB + k0 + 16);
    v16h av, bv;
#pragma unroll
    for (int i = 0; i < 8; ++i) {
      av[i] = a0[i]; av[8 + i] = a1[i];
      bv[i] = b0[i]; bv[8 + i] = b1[i];
    }
    acc = __builtin_amdgcn_wmma_f32_16x16x32_f16(
        /*neg_a=*/false, av, /*neg_b=*/false, bv,
        /*c_mod=*/(short)0, acc, /*reuse_a=*/false, /*reuse_b=*/false);
  }
  const float inv = misc[0];
  const int col = n0 + (lane & 15);
  const float dj = dvec[col];
  const int rbase = m0 + ((lane < 16) ? 0 : 8);
#pragma unroll
  for (int e = 0; e < 8; ++e) {
    int row = rbase + e;
    if (row < BATCH)
      R[(size_t)row * NOUT + col] = fmaxf(inv * acc[e] + dj, 0.f);
  }
}

// --------------------------- MLP head: relu(fc1) -> fc2 -> log_softmax -----
__global__ void k_head(const float* __restrict__ R,
                       const float* __restrict__ fc1_w, const float* __restrict__ fc1_b,
                       const float* __restrict__ out_w, const float* __restrict__ out_b,
                       float* __restrict__ out)
{
  __shared__ float w1[64 * 128];
  __shared__ float wo[10 * 64];
  __shared__ float b1s[64], bos[10];
  const int t = threadIdx.x;
  for (int i = t; i < 64 * 128; i += 256) w1[i] = fc1_w[i];
  for (int i = t; i < 640; i += 256) wo[i] = out_w[i];
  if (t < 64) b1s[t] = fc1_b[t];
  if (t < 10) bos[t] = out_b[t];
  __syncthreads();
  const int b = blockIdx.x * 256 + t;
  if (b >= BATCH) return;
  const float* rb = R + (size_t)b * NOUT;
  float r1[64];
  for (int o = 0; o < 64; ++o) {
    float s = b1s[o];
#pragma unroll 8
    for (int k = 0; k < 128; ++k) s += rb[k] * w1[o * 128 + k];
    r1[o] = fmaxf(s, 0.f);
  }
  float lg[10];
#pragma unroll
  for (int o = 0; o < 10; ++o) {
    float s = bos[o];
#pragma unroll 8
    for (int k = 0; k < 64; ++k) s += r1[k] * wo[o * 64 + k];
    lg[o] = s;
  }
  float mx = lg[0];
#pragma unroll
  for (int o = 1; o < 10; ++o) mx = fmaxf(mx, lg[o]);
  float se = 0.f;
#pragma unroll
  for (int o = 0; o < 10; ++o) se += expf(lg[o] - mx);
  const float lse = mx + logf(se);
#pragma unroll
  for (int o = 0; o < 10; ++o) out[(size_t)b * 10 + o] = lg[o] - lse;
}

// ---------------------------------------------------------------------------
extern "C" void kernel_launch(void* const* d_in, const int* in_sizes, int n_in,
                              void* d_out, int out_size, void* d_ws, size_t ws_size,
                              hipStream_t stream)
{
  (void)in_sizes; (void)n_in; (void)out_size; (void)ws_size;
  const float* x        = (const float*)d_in[0];
  const int*   assign   = (const int*)  d_in[1];
  const float* h_mat    = (const float*)d_in[2];
  const float* b_mat    = (const float*)d_in[3];
  const float* a_arr    = (const float*)d_in[4];
  const float* noise    = (const float*)d_in[5];
  const float* conv_w   = (const float*)d_in[6];
  const float* conv_b   = (const float*)d_in[7];
  const float* bn_g     = (const float*)d_in[8];
  const float* bn_b     = (const float*)d_in[9];
  const float* cutW_big = (const float*)d_in[10];
  const float* cutW_sm  = (const float*)d_in[11];
  const float* fc1_w    = (const float*)d_in[12];
  const float* fc1_b    = (const float*)d_in[13];
  const float* out_w    = (const float*)d_in[14];
  const float* out_b    = (const float*)d_in[15];
  float* out = (float*)d_out;

  char* w = (char*)d_ws;
  auto carve = [&](size_t bytes) -> void* {
    void* p = (void*)w; w += (bytes + 255) & ~(size_t)255; return p;
  };
  _Float16* A     = (_Float16*)carve((size_t)MPAD * KTOT * 2); // 14.8 MB
  _Float16* Wall  = (_Float16*)carve((size_t)NOUT * KTOT * 2); //  1.0 MB
  float* R          = (float*)carve((size_t)BATCH * NOUT * 4); //  0.9 MB
  float* chanstats  = (float*)carve(80 * 4);
  float* scaleshift = (float*)carve(80 * 4);
  float* bstats     = (float*)carve(20 * 4);
  float* dvec       = (float*)carve(128 * 4);
  float* misc       = (float*)carve(64);

  k_init<<<128, 256, 0, stream>>>(chanstats, bstats, A);
  // pass 0: conv -> per-channel stats
  k_conv<32><<<dim3(BATCH, 2), 256, 0, stream>>>(x, conv_w, conv_b, scaleshift,
                                                 chanstats, bstats, A, 0);
  k_conv<16><<<dim3(BATCH, 8), 256, 0, stream>>>(x, conv_w, conv_b, scaleshift,
                                                 chanstats, bstats, A, 0);
  k_chanfin<<<1, 64, 0, stream>>>(chanstats, bn_g, bn_b, scaleshift);
  // pass 1: conv -> BN -> relu -> pool -> f16 A + branch stats
  k_conv<32><<<dim3(BATCH, 2), 256, 0, stream>>>(x, conv_w, conv_b, scaleshift,
                                                 chanstats, bstats, A, 1);
  k_conv<16><<<dim3(BATCH, 8), 256, 0, stream>>>(x, conv_w, conv_b, scaleshift,
                                                 chanstats, bstats, A, 1);
  k_weights<<<128, 256, 0, stream>>>(assign, h_mat, b_mat, a_arr, noise,
                                     cutW_big, cutW_sm, bstats, Wall, dvec, misc);
  k_gemm<<<MPAD / 16, 256, 0, stream>>>(A, Wall, dvec, misc, R);
  k_head<<<(BATCH + 255) / 256, 256, 0, stream>>>(R, fc1_w, fc1_b, out_w, out_b, out);
}
